// Autoformer_31490700214721
// MI455X (gfx1250) — matrix operand order repaired and checked by reading, not compiled, not run on hardware
//
#include <hip/hip_runtime.h>
#include <math.h>

// ---------------------------------------------------------------------------
// Autoformer forward on MI455X (gfx1250), wave32 + WMMA bf16.
// All GEMMs: bf16 A/B, fp32 accumulate via v_wmma_f32_16x16x32_bf16.
// All GEMMs are NT: weights are transposed to (N x K) bf16 once per launch so
// both A and B tiles stage with contiguous 128-bit loads when K % 32 == 0.
// FFT autocorrelation replaced by batched Q·K^T GEMM + circular-diagonal sum.
// Requires ~850MB of workspace (activations + bf16 weight copies).
// Input order assumed = setup_inputs() dict insertion order, recursively.
// ---------------------------------------------------------------------------

typedef __attribute__((ext_vector_type(16))) __bf16 v16bf;
typedef __attribute__((ext_vector_type(8)))  float  v8f;

#define BM 128
#define BN 64
#define BK 32
#define PAD 8              // LDS row stride = BK+PAD halves = 80 B (16B-aligned)

__device__ __forceinline__ unsigned short f2bf(float f) {
  unsigned int u = __float_as_uint(f);
  u += 0x7FFFu + ((u >> 16) & 1u);          // round-to-nearest-even
  return (unsigned short)(u >> 16);
}
__device__ __forceinline__ float bf2f(unsigned short h) {
  return __uint_as_float(((unsigned int)h) << 16);
}
__device__ __forceinline__ float gelu_exact(float v) {
  return 0.5f * v * (1.0f + erff(v * 0.70710678118654752f));
}

union Frag { v16bf v; unsigned int u[8]; };

// ---------------------------------------------------------------------------
// GEMM (NT): C[M,N] = A[M,K](bf16,row-major) x Bt[N,K](bf16,row-major)^T + bias
// flags: bit0 = exact GELU epilogue. Cf (fp32) and/or Cb (bf16) outputs.
// ---------------------------------------------------------------------------
__global__ __launch_bounds__(256) void k_gemm(
    const unsigned short* __restrict__ A,
    const unsigned short* __restrict__ Bt,
    const float* __restrict__ bias,
    float* __restrict__ Cf,
    unsigned short* __restrict__ Cb,
    int M, int N, int K, int flags)
{
  __shared__ __align__(16) unsigned short As[BM][BK + PAD];
  __shared__ __align__(16) unsigned short Bs[BN][BK + PAD];
  const int tid  = threadIdx.x;
  const int bm   = blockIdx.x * BM;
  const int bn   = blockIdx.y * BN;
  const int lane = tid & 31;
  const int wave = tid >> 5;
  const int wm   = (wave & 3) << 5;   // 4 waves along M
  const int wn   = (wave >> 2) << 5;  // 2 waves along N
  const int r    = lane & 15;
  const int half = lane >> 4;
  const bool kaligned = (K % BK) == 0;

  v8f z8 = {0.f,0.f,0.f,0.f,0.f,0.f,0.f,0.f};
  v8f acc[2][2];
  acc[0][0] = z8; acc[0][1] = z8; acc[1][0] = z8; acc[1][1] = z8;

  for (int kb = 0; kb < K; kb += BK) {
    if (kaligned) {
      // ---- vector staging: A = two b128, B = one b128 per thread ----
      {
        int row = tid >> 1, c0 = (tid & 1) * 16;
        const uint4* src = (const uint4*)(A + (long)(bm + row) * K + kb + c0);
        uint4 d0 = src[0];
        uint4 d1 = src[1];
        if (kb + BK < K)
          __builtin_prefetch(A + (long)(bm + row) * K + kb + BK + c0);
        *(uint4*)&As[row][c0]     = d0;
        *(uint4*)&As[row][c0 + 8] = d1;
      }
      {
        int n = tid >> 2, k0 = (tid & 3) * 8;
        int gn = bn + n;
        uint4 bv = {0u, 0u, 0u, 0u};
        if (gn < N) {
          bv = *(const uint4*)(Bt + (long)gn * K + kb + k0);
          if (kb + BK < K)
            __builtin_prefetch(Bt + (long)gn * K + kb + BK + k0);
        }
        *(uint4*)&Bs[n][k0] = bv;
      }
    } else {
      // ---- scalar guarded staging (K == 21 embeddings only) ----
      for (int i = tid; i < BM * BK; i += 256) {
        int row = i >> 5, col = i & 31;
        int gk = kb + col;
        As[row][col] = (gk < K) ? A[(long)(bm + row) * K + gk] : (unsigned short)0;
      }
      for (int i = tid; i < BN * BK; i += 256) {
        int n = i >> 5, k = i & 31;
        int gn = bn + n, gk = kb + k;
        Bs[n][k] = (gn < N && gk < K) ? Bt[(long)gn * K + gk] : (unsigned short)0;
      }
    }
    __syncthreads();

    Frag a[2], b[2];
#pragma unroll
    for (int f = 0; f < 2; ++f) {
      int arow = wm + f * 16 + r;
#pragma unroll
      for (int j = 0; j < 8; ++j) {       // ISA 16-bit A layout: lanes0-15 K=0..7,16..23
        int kp = ((j < 4) ? (2 * j) : (2 * j + 8)) + 8 * half;
        a[f].u[j] = *(const unsigned int*)&As[arow][kp];
      }
      int bcol = wn + f * 16 + r;
#pragma unroll
      for (int j = 0; j < 8; ++j) {       // B layout: lane=N, lanes0-15 K=0..7,16..23 (same as A)
        int kp = ((j < 4) ? (2 * j) : (2 * j + 8)) + 8 * half;
        b[f].u[j] = *(const unsigned int*)&Bs[bcol][kp];
      }
    }
#pragma unroll
    for (int fm = 0; fm < 2; ++fm)
#pragma unroll
      for (int fn = 0; fn < 2; ++fn)
        acc[fm][fn] = __builtin_amdgcn_wmma_f32_16x16x32_bf16(
            false, a[fm].v, false, b[fn].v, (short)0, acc[fm][fn], false, false);
    __syncthreads();
  }

#pragma unroll
  for (int fm = 0; fm < 2; ++fm)
#pragma unroll
    for (int fn = 0; fn < 2; ++fn) {
      int ccol = bn + wn + fn * 16 + r;
      if (ccol >= N) continue;
      float bv = bias ? bias[ccol] : 0.f;
#pragma unroll
      for (int j = 0; j < 8; ++j) {
        int crow = bm + wm + fm * 16 + (half << 3) + j;   // C layout: VGPR j -> M=j+8*half
        if (crow >= M) continue;
        float v = acc[fm][fn][j] + bv;
        if (flags & 1) v = gelu_exact(v);
        long idx = (long)crow * N + ccol;
        if (Cf) Cf[idx] = v;
        if (Cb) Cb[idx] = f2bf(v);
      }
    }
}

// ---------------------------------------------------------------------------
// Batched correlation: S = Q·K^T per batch, accumulate circular diagonals
// corr[b,tau] += sum_{i-j == tau (mod L)} S[i,j].   (A and B both L x K, NT)
// K == 512 (always BK-aligned) -> vector staging unconditionally.
// ---------------------------------------------------------------------------
__global__ __launch_bounds__(256) void k_corr(
    const unsigned short* __restrict__ Qm,
    const unsigned short* __restrict__ Km,
    float* __restrict__ corr, int L, int K)
{
  __shared__ __align__(16) unsigned short As[BM][BK + PAD];
  __shared__ __align__(16) unsigned short Bs[BN][BK + PAD];
  __shared__ float diag[BM + BN];
  const int tid  = threadIdx.x;
  const int b    = blockIdx.z;
  const unsigned short* A  = Qm + (long)b * L * K;
  const unsigned short* Bt = Km + (long)b * L * K;
  const int bm = blockIdx.x * BM;
  const int bn = blockIdx.y * BN;
  const int lane = tid & 31;
  const int wave = tid >> 5;
  const int wm = (wave & 3) << 5;
  const int wn = (wave >> 2) << 5;
  const int r = lane & 15;
  const int half = lane >> 4;

  v8f z8 = {0.f,0.f,0.f,0.f,0.f,0.f,0.f,0.f};
  v8f acc[2][2];
  acc[0][0] = z8; acc[0][1] = z8; acc[1][0] = z8; acc[1][1] = z8;
  for (int i = tid; i < BM + BN; i += 256) diag[i] = 0.f;

  for (int kb = 0; kb < K; kb += BK) {
    {
      int row = tid >> 1, c0 = (tid & 1) * 16;
      const uint4* src = (const uint4*)(A + (long)(bm + row) * K + kb + c0);
      uint4 d0 = src[0];
      uint4 d1 = src[1];
      if (kb + BK < K)
        __builtin_prefetch(A + (long)(bm + row) * K + kb + BK + c0);
      *(uint4*)&As[row][c0]     = d0;
      *(uint4*)&As[row][c0 + 8] = d1;
    }
    {
      int n = tid >> 2, k0 = (tid & 3) * 8;
      uint4 bv = *(const uint4*)(Bt + (long)(bn + n) * K + kb + k0);
      if (kb + BK < K)
        __builtin_prefetch(Bt + (long)(bn + n) * K + kb + BK + k0);
      *(uint4*)&Bs[n][k0] = bv;
    }
    __syncthreads();

    Frag a[2], bb[2];
#pragma unroll
    for (int f = 0; f < 2; ++f) {
      int arow = wm + f * 16 + r;
#pragma unroll
      for (int j = 0; j < 8; ++j) {
        int kp = ((j < 4) ? (2 * j) : (2 * j + 8)) + 8 * half;
        a[f].u[j] = *(const unsigned int*)&As[arow][kp];
      }
      int bcol = wn + f * 16 + r;
#pragma unroll
      for (int j = 0; j < 8; ++j) {
        int kp = ((j < 4) ? (2 * j) : (2 * j + 8)) + 8 * half;
        bb[f].u[j] = *(const unsigned int*)&Bs[bcol][kp];
      }
    }
#pragma unroll
    for (int fm = 0; fm < 2; ++fm)
#pragma unroll
      for (int fn = 0; fn < 2; ++fn)
        acc[fm][fn] = __builtin_amdgcn_wmma_f32_16x16x32_bf16(
            false, a[fm].v, false, bb[fn].v, (short)0, acc[fm][fn], false, false);
    __syncthreads();
  }

#pragma unroll
  for (int fm = 0; fm < 2; ++fm)
#pragma unroll
    for (int fn = 0; fn < 2; ++fn) {
      int jl = wn + fn * 16 + r;
#pragma unroll
      for (int j = 0; j < 8; ++j) {
        int il = wm + fm * 16 + (half << 3) + j;
        atomicAdd(&diag[jl - il + (BM - 1)], acc[fm][fn][j]);   // ds_add_f32
      }
    }
  __syncthreads();
  for (int dd = tid; dd < BM + BN - 1; dd += 256) {
    int tau = (bm - bn + (BM - 1) - dd) % L;
    if (tau < 0) tau += L;
    atomicAdd(&corr[(long)b * L + tau], diag[dd]);
  }
}

// ---------------------------------------------------------------------------
// Small elementwise / reduction kernels
// ---------------------------------------------------------------------------
__global__ void k_f2b(const float* __restrict__ s, unsigned short* __restrict__ d, long n) {
  long i = (long)blockIdx.x * blockDim.x + threadIdx.x;
  if (i < n) d[i] = f2bf(s[i]);
}
// transpose-convert: src (K x N fp32, row-major) -> dst (N x K bf16, row-major)
__global__ void k_f2bt(const float* __restrict__ s, unsigned short* __restrict__ d,
                       int N, int K) {
  long i = (long)blockIdx.x * blockDim.x + threadIdx.x;
  if (i >= (long)N * K) return;
  int n = (int)(i / K);
  int k = (int)(i % K);
  d[i] = f2bf(s[(long)k * N + n]);
}
__global__ void k_zero(float* __restrict__ p, long n) {
  long i = (long)blockIdx.x * blockDim.x + threadIdx.x;
  if (i < n) p[i] = 0.f;
}

// series_decomp (kernel=25, edge pad) on the tail 768 rows; fills trend[:, :768]
// and seasonal (bf16) for all 1536 decoder rows (second half zero).
__global__ void k_decomp(const float* __restrict__ x, float* __restrict__ trend,
                         unsigned short* __restrict__ seasb,
                         int Bc, int L, int C, int P) {
  long i = (long)blockIdx.x * blockDim.x + threadIdx.x;
  long total = (long)Bc * L * C;
  if (i >= total) return;
  int c = (int)(i % C);
  long bt = i / C;
  int t = (int)(bt % L);
  int b = (int)(bt / L);
  if (t >= P) { seasb[i] = 0; return; }
  int ts = P + t;                               // source row in [768,1536)
  float s = 0.f;
  for (int j = -12; j <= 12; ++j) {
    int tt = ts + j;
    tt = tt < 0 ? 0 : (tt > L - 1 ? L - 1 : tt);
    s += x[((long)b * L + tt) * C + c];
  }
  float avg = s * (1.f / 25.f);
  trend[i] = avg;
  seasb[i] = f2bf(x[((long)b * L + ts) * C + c] - avg);
}

__global__ void k_trendmean(float* __restrict__ trend, int Bc, int L, int C, int P) {
  int i = blockIdx.x * blockDim.x + threadIdx.x;
  if (i >= Bc * C) return;
  int c = i % C, b = i / C;
  float s = 0.f;
  for (int t = 0; t < P; ++t) s += trend[((long)b * L + t) * C + c];
  s /= (float)P;
  for (int t = P; t < L; ++t) trend[((long)b * L + t) * C + c] = s;
}

// top-2 over batch-mean of corr, then per-batch 2-way softmax weights.
__global__ __launch_bounds__(256) void k_topk(const float* __restrict__ corr,
                                              int Bc, int L, float scale,
                                              int* __restrict__ idx_out,
                                              float* __restrict__ w_out) {
  __shared__ float sv[512];
  __shared__ int   si[512];
  __shared__ int   sidx[2];
  int tid = threadIdx.x;
  float b0 = -3.4e38f, b1 = -3.4e38f; int i0 = -1, i1 = -1;
  for (int t = tid; t < L; t += 256) {
    float s = 0.f;
    for (int b = 0; b < Bc; ++b) s += corr[(long)b * L + t];
    if (s > b0) { b1 = b0; i1 = i0; b0 = s; i0 = t; }
    else if (s > b1) { b1 = s; i1 = t; }
  }
  sv[tid] = b0; si[tid] = i0; sv[256 + tid] = b1; si[256 + tid] = i1;
  __syncthreads();
  if (tid == 0) {
    float m0 = -3.4e38f, m1 = -3.4e38f; int j0 = 0, j1 = 0;
    for (int i = 0; i < 512; ++i) {
      int ii = si[i]; if (ii < 0) continue;
      float v = sv[i];
      if (v > m0)      { m1 = m0; j1 = j0; m0 = v; j0 = ii; }
      else if (v > m1) { m1 = v; j1 = ii; }
    }
    sidx[0] = j0; sidx[1] = j1;
    idx_out[0] = j0; idx_out[1] = j1;
  }
  __syncthreads();
  if (tid < Bc) {
    float v0 = corr[(long)tid * L + sidx[0]] * scale;
    float v1 = corr[(long)tid * L + sidx[1]] * scale;
    float m = fmaxf(v0, v1);
    float e0 = expf(v0 - m), e1 = expf(v1 - m);
    float s = e0 + e1;
    w_out[tid * 2 + 0] = e0 / s;
    w_out[tid * 2 + 1] = e1 / s;
  }
}

// agg[b,l,h,e] = w0*v[b,l,(h+i0)%8,e] + w1*v[b,l,(h+i1)%8,e]
__global__ void k_agg(const unsigned short* __restrict__ vb, const int* __restrict__ idx,
                      const float* __restrict__ w, unsigned short* __restrict__ out,
                      long total, int L, int D) {
  long i = (long)blockIdx.x * blockDim.x + threadIdx.x;
  if (i >= total) return;
  int d = (int)(i % D);
  long bl = i / D;
  int b = (int)(bl / L);
  int h = d >> 6, e = d & 63;
  int h0 = (h + idx[0]) & 7;
  int h1 = (h + idx[1]) & 7;
  float v0 = bf2f(vb[bl * D + (h0 << 6) + e]);
  float v1 = bf2f(vb[bl * D + (h1 << 6) + e]);
  out[i] = f2bf(w[b * 2] * v0 + w[b * 2 + 1] * v1);
}

// xo = LN(x (+res)); also emits bf16 copy. One block per row, D==512.
__global__ __launch_bounds__(256) void k_ln(const float* __restrict__ x,
                                            const float* __restrict__ res,
                                            const float* __restrict__ g,
                                            const float* __restrict__ bt,
                                            float* __restrict__ xo,
                                            unsigned short* __restrict__ xb, int D) {
  long row = blockIdx.x;
  int tid = threadIdx.x;
  const float* xr = x + row * (long)D;
  float v0 = xr[tid], v1 = xr[tid + 256];
  if (res) {
    v0 += res[row * (long)D + tid];
    v1 += res[row * (long)D + tid + 256];
  }
  __shared__ float sbuf[256];
  sbuf[tid] = v0 + v1;
  __syncthreads();
  for (int off = 128; off > 0; off >>= 1) { if (tid < off) sbuf[tid] += sbuf[tid + off]; __syncthreads(); }
  float mean = sbuf[0] / (float)D;
  __syncthreads();
  sbuf[tid] = v0 * v0 + v1 * v1;
  __syncthreads();
  for (int off = 128; off > 0; off >>= 1) { if (tid < off) sbuf[tid] += sbuf[tid + off]; __syncthreads(); }
  float var = sbuf[0] / (float)D - mean * mean;
  float rstd = rsqrtf(var + 1e-5f);
  float o0 = (v0 - mean) * rstd * g[tid] + bt[tid];
  float o1 = (v1 - mean) * rstd * g[tid + 256] + bt[tid + 256];
  if (xo) { xo[row * (long)D + tid] = o0; xo[row * (long)D + tid + 256] = o1; }
  if (xb) { xb[row * (long)D + tid] = f2bf(o0); xb[row * (long)D + tid + 256] = f2bf(o1); }
}

// decoder residual conv: trend[b,l,o] += sum_c sum_t x[b,clamp(l+t-1),c]*W[o,c,t]
__global__ void k_convres(const float* __restrict__ x, const float* __restrict__ W,
                          float* __restrict__ trend, int Bc, int L, int C, int D) {
  long i = (long)blockIdx.x * blockDim.x + threadIdx.x;
  long total = (long)Bc * L * C;
  if (i >= total) return;
  int o = (int)(i % C);
  long bl = i / C;
  int l = (int)(bl % L);
  int b = (int)(bl / L);
  const float* xm = x + ((long)b * L + (l > 0 ? l - 1 : 0)) * D;
  const float* x0 = x + ((long)b * L + l) * D;
  const float* xp = x + ((long)b * L + (l < L - 1 ? l + 1 : L - 1)) * D;
  float s = 0.f;
  for (int c = 0; c < D; ++c) {
    const float* wr = W + ((long)o * D + c) * 3;
    s += xm[c] * wr[0] + x0[c] * wr[1] + xp[c] * wr[2];
  }
  trend[i] += s;
}

__global__ void k_final(const float* __restrict__ trend, const float* __restrict__ proj,
                        float* __restrict__ out, int Bc, int L, int C, int P) {
  long i = (long)blockIdx.x * blockDim.x + threadIdx.x;
  long total = (long)Bc * P * C;
  if (i >= total) return;
  int c = (int)(i % C);
  long bt = i / C;
  int t = (int)(bt % P);
  int b = (int)(bt / P);
  long src = ((long)b * L + (L - P) + t) * C + c;
  out[i] = trend[src] + proj[src];
}

// ---------------------------------------------------------------------------
// Host orchestration
// ---------------------------------------------------------------------------
extern "C" void kernel_launch(void* const* d_in, const int* in_sizes, int n_in,
                              void* d_out, int out_size, void* d_ws, size_t ws_size,
                              hipStream_t stream) {
  (void)in_sizes; (void)n_in; (void)out_size; (void)ws_size;
  const int Bc = 32, L = 1536, D = 512, FF = 2048, C = 21, P = 768;
  const long M = (long)Bc * L;  // 49152

  auto Fp = [&](int i) { return (const float*)d_in[i]; };

  // ---- workspace layout -------------------------------------------------
  char* base = (char*)d_ws;
  size_t off = 0;
  auto alloc = [&](size_t bytes) -> void* {
    void* p = base + off;
    off = (off + bytes + 255) & ~(size_t)255;
    return p;
  };
  float* enc  = (float*)alloc(M * D * sizeof(float));
  float* dec  = (float*)alloc(M * D * sizeof(float));
  float* tmp  = (float*)alloc(M * D * sizeof(float));
  unsigned short* xb    = (unsigned short*)alloc(M * D * 2);
  unsigned short* qb    = (unsigned short*)alloc(M * D * 2);
  unsigned short* kb    = (unsigned short*)alloc(M * D * 2);
  unsigned short* vb    = (unsigned short*)alloc(M * D * 2);
  unsigned short* aggb  = (unsigned short*)alloc(M * D * 2);
  unsigned short* encb  = (unsigned short*)alloc(M * D * 2);
  unsigned short* hb    = (unsigned short*)alloc(M * FF * 2);
  float* corr = (float*)alloc((long)Bc * L * sizeof(float));
  int*   idxb = (int*)alloc(2 * sizeof(int));
  float* wsm  = (float*)alloc(Bc * 2 * sizeof(float));
  float* trend = (float*)alloc((long)Bc * L * C * sizeof(float));
  unsigned short* seasb = (unsigned short*)alloc(M * C * 2);
  unsigned short* xencb = (unsigned short*)alloc(M * C * 2);

  const size_t SZA = 262144, SZF = 1048576, SZE = 10752;   // 512*512, 512*2048, 21*512
  const size_t ENC_STRIDE = 4 * SZA + 2 * SZF;
  const size_t o_embE = 0, o_embD = SZE;
  size_t o_encL[2] = {2 * SZE, 2 * SZE + ENC_STRIDE};
  const size_t o_dec  = 2 * SZE + 2 * ENC_STRIDE;
  const size_t o_proj = o_dec + 8 * SZA + 2 * SZF;
  unsigned short* wts = (unsigned short*)alloc((o_proj + SZE) * 2);

  // transpose-convert weight src (Kw x Nw) -> bf16 (Nw x Kw)
  auto cvtT = [&](int idx, size_t dst, int Nw, int Kw) {
    long n = (long)Nw * Kw;
    k_f2bt<<<(int)((n + 255) / 256), 256, 0, stream>>>((const float*)d_in[idx], wts + dst, Nw, Kw);
  };
  auto gemm = [&](const unsigned short* A, const unsigned short* Bt, const float* bias,
                  float* Cf, unsigned short* Cb, int Mr, int Nr, int Kr, int flags) {
    dim3 g(Mr / BM, (Nr + BN - 1) / BN);
    k_gemm<<<g, 256, 0, stream>>>(A, Bt, bias, Cf, Cb, Mr, Nr, Kr, flags);
  };
  auto attn = [&](const unsigned short* qsrc, const unsigned short* kvsrc,
                  const unsigned short* Wq, const float* bq,
                  const unsigned short* Wk, const float* bk2,
                  const unsigned short* Wv, const float* bv2,
                  const unsigned short* Wo, const float* bo,
                  float* x, const float* ng, const float* nb, unsigned short* xbo) {
    gemm(qsrc, Wq, bq,  nullptr, qb, (int)M, D, D, 0);
    gemm(kvsrc, Wk, bk2, nullptr, kb, (int)M, D, D, 0);
    gemm(kvsrc, Wv, bv2, nullptr, vb, (int)M, D, D, 0);
    k_zero<<<(Bc * L + 255) / 256, 256, 0, stream>>>(corr, (long)Bc * L);
    k_corr<<<dim3(L / BM, L / BN, Bc), 256, 0, stream>>>(qb, kb, corr, L, D);
    k_topk<<<1, 256, 0, stream>>>(corr, Bc, L, 1.f / 512.f, idxb, wsm);
    long tot = M * D;
    k_agg<<<(int)((tot + 255) / 256), 256, 0, stream>>>(vb, idxb, wsm, aggb, tot, L, D);
    gemm(aggb, Wo, bo, tmp, nullptr, (int)M, D, D, 0);
    k_ln<<<(int)M, 256, 0, stream>>>(x, tmp, ng, nb, x, xbo, D);
  };
  auto ffn = [&](float* x, unsigned short* xb_, const unsigned short* Wc1, const float* bc1,
                 const unsigned short* Wc2, const float* bc2, const float* ng, const float* nb) {
    gemm(xb_, Wc1, bc1, nullptr, hb, (int)M, FF, D, 1 /*gelu*/);
    gemm(hb, Wc2, bc2, tmp, nullptr, (int)M, D, FF, 0);
    k_ln<<<(int)M, 256, 0, stream>>>(x, tmp, ng, nb, x, xb_, D);
  };

  // ---- weight conversion (transposed) to bf16 ---------------------------
  cvtT(1, o_embE, D, C);                        // enc_emb.W (21x512) -> (512x21)
  cvtT(3, o_embD, D, C);                        // dec_emb.W
  for (int l = 0; l < 2; ++l) {
    int pi = 5 + l * 16;
    size_t ob = o_encL[l];
    cvtT(pi + 0, ob + 0 * SZA, D, D);           // Wq
    cvtT(pi + 2, ob + 1 * SZA, D, D);           // Wk
    cvtT(pi + 4, ob + 2 * SZA, D, D);           // Wv
    cvtT(pi + 6, ob + 3 * SZA, D, D);           // Wo
    cvtT(pi + 8, ob + 4 * SZA, FF, D);          // Wc1 (512x2048) -> (2048x512)
    cvtT(pi + 10, ob + 4 * SZA + SZF, D, FF);   // Wc2 (2048x512) -> (512x2048)
  }
  cvtT(39, o_dec + 0 * SZA, D, D); cvtT(41, o_dec + 1 * SZA, D, D);
  cvtT(43, o_dec + 2 * SZA, D, D); cvtT(45, o_dec + 3 * SZA, D, D);
  cvtT(47, o_dec + 4 * SZA, D, D); cvtT(49, o_dec + 5 * SZA, D, D);
  cvtT(51, o_dec + 6 * SZA, D, D); cvtT(53, o_dec + 7 * SZA, D, D);
  cvtT(55, o_dec + 8 * SZA, FF, D); cvtT(57, o_dec + 8 * SZA + SZF, D, FF);
  cvtT(68, o_proj, C, D);                       // Wproj (512x21) -> (21x512)

  // ---- decomposition + trend init --------------------------------------
  k_f2b<<<(int)((M * C + 255) / 256), 256, 0, stream>>>(Fp(0), xencb, M * C);
  k_decomp<<<(int)((M * C + 255) / 256), 256, 0, stream>>>(Fp(0), trend, seasb, Bc, L, C, P);
  k_trendmean<<<(Bc * C + 255) / 256, 256, 0, stream>>>(trend, Bc, L, C, P);

  // ---- encoder -----------------------------------------------------------
  gemm(xencb, wts + o_embE, Fp(2), enc, nullptr, (int)M, D, C, 0);
  k_f2b<<<(int)((M * D + 255) / 256), 256, 0, stream>>>(enc, xb, M * D);
  for (int l = 0; l < 2; ++l) {
    int pi = 5 + l * 16;
    const unsigned short* Wq  = wts + o_encL[l];
    const unsigned short* Wk  = Wq + SZA;
    const unsigned short* Wv  = Wq + 2 * SZA;
    const unsigned short* Wo  = Wq + 3 * SZA;
    const unsigned short* Wc1 = Wq + 4 * SZA;
    const unsigned short* Wc2 = Wq + 4 * SZA + SZF;
    attn(xb, xb, Wq, Fp(pi + 1), Wk, Fp(pi + 3), Wv, Fp(pi + 5), Wo, Fp(pi + 7),
         enc, Fp(pi + 12), Fp(pi + 13), xb);
    ffn(enc, xb, Wc1, Fp(pi + 9), Wc2, Fp(pi + 11), Fp(pi + 14), Fp(pi + 15));
  }
  k_ln<<<(int)M, 256, 0, stream>>>(enc, nullptr, Fp(37), Fp(38), enc, encb, D);

  // ---- decoder -----------------------------------------------------------
  gemm(seasb, wts + o_embD, Fp(4), dec, nullptr, (int)M, D, C, 0);
  k_f2b<<<(int)((M * D + 255) / 256), 256, 0, stream>>>(dec, xb, M * D);
  {
    const unsigned short* sWq = wts + o_dec;
    const unsigned short* cWq = wts + o_dec + 4 * SZA;
    const unsigned short* Wc1 = wts + o_dec + 8 * SZA;
    const unsigned short* Wc2 = Wc1 + SZF;
    attn(xb, xb,  sWq, Fp(40), sWq + SZA, Fp(42), sWq + 2 * SZA, Fp(44), sWq + 3 * SZA, Fp(46),
         dec, Fp(59), Fp(60), xb);
    attn(xb, encb, cWq, Fp(48), cWq + SZA, Fp(50), cWq + 2 * SZA, Fp(52), cWq + 3 * SZA, Fp(54),
         dec, Fp(61), Fp(62), xb);
    ffn(dec, xb, Wc1, Fp(56), Wc2, Fp(58), Fp(63), Fp(64));
    k_convres<<<(int)((M * C + 255) / 256), 256, 0, stream>>>(dec, Fp(65), trend, Bc, L, C, D);
  }
  k_ln<<<(int)M, 256, 0, stream>>>(dec, nullptr, Fp(66), Fp(67), dec, xb, D);
  gemm(xb, wts + o_proj, Fp(69), tmp, nullptr, (int)M, C, D, 0);
  k_final<<<(int)(((long)Bc * P * C + 255) / 256), 256, 0, stream>>>(
      trend, tmp, (float*)d_out, Bc, L, C, P);
}
